// LWGA_Block_65180423685088
// MI455X (gfx1250) — matrix operand order repaired and checked by reading, compiled, not run on hardware
//
#include <hip/hip_runtime.h>

// ---------------------------------------------------------------------------
// CDNA5 (gfx1250) implementation of the LWGA block.
// GEMM-heavy paths (all 1x1 convs + the 3x3 LA conv as 9 shifted GEMM taps)
// run on v_wmma_f32_16x16x32_bf16 (wave32, f32 accum), register-tiled 2x2
// fragments per wave (4 WMMAs per k-chunk, b128 fragment loads, branch-free
// zero-page for conv padding). Activations staged NHWC bf16 via an LDS-tiled
// transpose so fragment K-pairs are contiguous. Small ops are scalar kernels.
// ---------------------------------------------------------------------------

typedef __attribute__((ext_vector_type(16))) __bf16 v16bf;
typedef __attribute__((ext_vector_type(8)))  __bf16 v8bf;
typedef __attribute__((ext_vector_type(8)))  float  v8f;

union BF16x16 { v16bf v; v8bf h[2]; };

__device__ __forceinline__ float sigmoidf_(float v) { return 1.f / (1.f + expf(-v)); }

// ---------------- zero fill ----------------
__global__ void zero_k(__bf16* p, int n) {
  int i = blockIdx.x * 256 + threadIdx.x;
  if (i < n) p[i] = (__bf16)0.f;
}

// ---------------- weight repack: f32 OIHW -> bf16 [tap][Opad][I], zero-padded
__global__ void pack_w(const float* __restrict__ w, __bf16* __restrict__ dst,
                       int O, int Opad, int I, int KH, int KW) {
  long i = (long)blockIdx.x * 256 + threadIdx.x;
  const long tot = (long)KH * KW * Opad * I;
  if (i >= tot) return;
  const int OI = Opad * I;
  const int t  = (int)(i / OI);        // tap = kh*KW + kw
  const int r  = (int)(i % OI);
  const int o  = r / I;
  const int ii = r % I;
  const int kh = t / KW, kw = t % KW;
  float v = 0.f;
  if (o < O) v = w[(((long)o * I + ii) * KH + kh) * KW + kw];
  dst[i] = (__bf16)v;
}

// ---------------- WMMA implicit-GEMM conv -----------------------------------
// D[b][m][hw] (f32 NCHW) = sum_{tap,k} A[tap][m][k] * Bx[b][hw+shift][k]
// A: bf16 [taps][Mpad][K]; Bx: bf16 NHWC [B][HW][K]; zpage: >=4KB of zeros.
// Wave tile: 32M x 32N (2x2 fragments, 4 accums). Block: 8 waves = 64M x 128N.
// Requires: K%32==0, W%16==0, HW%128==0, Mpad%64==0.
__global__ __launch_bounds__(256)
void wmma_conv_gemm(const __bf16* __restrict__ A,
                    const __bf16* __restrict__ Bx,
                    const __bf16* __restrict__ zpage,
                    float* __restrict__ D,
                    const float* __restrict__ bias,
                    const float* __restrict__ resid,
                    int M, int Mpad, int K, int Hh, int Ww,
                    int KH, int KW, int ph, int pw, int relu)
{
  const int lane = threadIdx.x & 31;
  const int wv   = threadIdx.x >> 5;
  const int mw   = wv & 1;           // 2 m-waves
  const int nw   = wv >> 1;          // 4 n-waves
  const int HW   = Hh * Ww;
  const int lm   = lane & 15;
  const int hi   = lane >> 4;

  const int  m0    = blockIdx.y * 64 + mw * 32;          // 32 rows for this wave
  const long nbase = (long)blockIdx.x * 128 + nw * 32;   // 32 cols (2 frags)
  const int  b     = (int)(nbase / HW);
  const int  hwb   = (int)(nbase % HW);

  const int h0f = hwb / Ww,        w0f = hwb % Ww + lm;
  const int h1f = (hwb + 16) / Ww, w1f = (hwb + 16) % Ww + lm;

  v8f acc[2][2];
  #pragma unroll
  for (int mf = 0; mf < 2; ++mf)
    #pragma unroll
    for (int nf = 0; nf < 2; ++nf)
      acc[mf][nf] = (v8f){0.f, 0.f, 0.f, 0.f, 0.f, 0.f, 0.f, 0.f};

  const int ka_off = hi ? 8 : 0;    // A frag: K {0..7,16..23} lo-half / {8..15,24..31} hi-half
  const int kb_off = hi ? 8 : 0;    // B frag: K {0..7,16..23} lo-half / {8..15,24..31} hi-half

  for (int kh = 0; kh < KH; ++kh) {
    for (int kw = 0; kw < KW; ++kw) {
      const int tap = kh * KW + kw;
      const __bf16* __restrict__ a0 = A + ((long)tap * Mpad + m0 + lm) * K;
      const __bf16* __restrict__ a1 = a0 + (long)16 * K;

      const int hs0 = h0f + kh - ph, ws0 = w0f + kw - pw;
      const int hs1 = h1f + kh - ph, ws1 = w1f + kw - pw;
      const bool ok0 = (hs0 >= 0) && (hs0 < Hh) && (ws0 >= 0) && (ws0 < Ww);
      const bool ok1 = (hs1 >= 0) && (hs1 < Hh) && (ws1 >= 0) && (ws1 < Ww);
      const __bf16* __restrict__ bp0 =
          ok0 ? (Bx + ((long)b * HW + (long)hs0 * Ww + ws0) * K) : zpage;
      const __bf16* __restrict__ bp1 =
          ok1 ? (Bx + ((long)b * HW + (long)hs1 * Ww + ws1) * K) : zpage;

      for (int kc = 0; kc < K; kc += 32) {
        BF16x16 A0, A1, B0, B1;
        A0.h[0] = *(const v8bf*)(a0 + kc + ka_off);
        A0.h[1] = *(const v8bf*)(a0 + kc + ka_off + 16);
        A1.h[0] = *(const v8bf*)(a1 + kc + ka_off);
        A1.h[1] = *(const v8bf*)(a1 + kc + ka_off + 16);
        B0.h[0] = *(const v8bf*)(bp0 + kc + kb_off);
        B0.h[1] = *(const v8bf*)(bp0 + kc + kb_off + 16);
        B1.h[0] = *(const v8bf*)(bp1 + kc + kb_off);
        B1.h[1] = *(const v8bf*)(bp1 + kc + kb_off + 16);
        acc[0][0] = __builtin_amdgcn_wmma_f32_16x16x32_bf16(false, A0.v, false, B0.v,
                                                            (short)0, acc[0][0], false, false);
        acc[0][1] = __builtin_amdgcn_wmma_f32_16x16x32_bf16(false, A0.v, false, B1.v,
                                                            (short)0, acc[0][1], false, false);
        acc[1][0] = __builtin_amdgcn_wmma_f32_16x16x32_bf16(false, A1.v, false, B0.v,
                                                            (short)0, acc[1][0], false, false);
        acc[1][1] = __builtin_amdgcn_wmma_f32_16x16x32_bf16(false, A1.v, false, B1.v,
                                                            (short)0, acc[1][1], false, false);
      }
    }
  }

  // C/D layout: VGPR v -> M = base + v + hi*8, N = lane % 16.
  const int hw0 = h0f * Ww + w0f;
  const int hw1 = h1f * Ww + w1f;
  #pragma unroll
  for (int mf = 0; mf < 2; ++mf) {
    #pragma unroll
    for (int v = 0; v < 8; ++v) {
      const int m = m0 + mf * 16 + hi * 8 + v;
      if (m < M) {
        const float bv = bias ? bias[m] : 0.f;
        const long rowb = ((long)b * M + m) * HW;
        #pragma unroll
        for (int nf = 0; nf < 2; ++nf) {
          const long off = rowb + (nf ? hw1 : hw0);
          float val = acc[mf][nf][v] + bv;
          if (resid) val += resid[off];
          if (relu)  val = fmaxf(val, 0.f);
          D[off] = val;
        }
      }
    }
  }
}

// ---------------- BatchNorm batch statistics --------------------------------
__global__ void bn_stats_k(const float* __restrict__ x, float* __restrict__ stats,
                           int C, int HW, int Bn) {
  const int c = blockIdx.x;
  __shared__ float ssum[256], ssq[256];
  float s = 0.f, q = 0.f;
  const long N = (long)Bn * HW;
  for (long i = threadIdx.x; i < N; i += 256) {
    const int b  = (int)(i / HW);
    const int hw = (int)(i % HW);
    const float v = x[((long)b * C + c) * HW + hw];
    s += v; q += v * v;
  }
  ssum[threadIdx.x] = s; ssq[threadIdx.x] = q;
  __syncthreads();
  for (int st = 128; st > 0; st >>= 1) {
    if (threadIdx.x < st) {
      ssum[threadIdx.x] += ssum[threadIdx.x + st];
      ssq[threadIdx.x]  += ssq[threadIdx.x + st];
    }
    __syncthreads();
  }
  if (threadIdx.x == 0) {
    const float mean = ssum[0] / (float)N;
    const float var  = ssq[0] / (float)N - mean * mean;
    stats[2 * c]     = mean;
    stats[2 * c + 1] = rsqrtf(var + 1e-5f);
  }
}

// ---------------- NCHW f32 (slice) -> NHWC bf16, LDS-tiled transpose --------
// Optional fused BN+ReLU. grid = (HW/32, C/32, B), block = 256 (32x8).
__global__ __launch_bounds__(256)
void to_nhwc_bf16(const float* __restrict__ src, __bf16* __restrict__ dst,
                  const float* __restrict__ stats,
                  int C, int Ctot, int c0, int HW) {
  __shared__ float tile[32][33];
  const int b   = blockIdx.z;
  const int hw0 = blockIdx.x * 32;
  const int ct0 = blockIdx.y * 32;            // channel tile base (within C)
  const int tx  = threadIdx.x & 31;
  const int ty  = threadIdx.x >> 5;           // 0..7
  #pragma unroll
  for (int r = 0; r < 32; r += 8) {
    const int cc = ct0 + ty + r;              // BN channel
    float v = src[((long)b * Ctot + c0 + cc) * HW + hw0 + tx];  // coalesced in hw
    if (stats) {
      v = (v - stats[2 * cc]) * stats[2 * cc + 1];
      v = v > 0.f ? v : 0.f;
    }
    tile[ty + r][tx] = v;
  }
  __syncthreads();
  #pragma unroll
  for (int r = 0; r < 32; r += 8) {
    const int hw = hw0 + ty + r;
    dst[((long)b * HW + hw) * C + ct0 + tx] = (__bf16)tile[tx][ty + r];  // coalesced in c
  }
}

// ---------------- BN+ReLU into an NCHW f32 channel slice --------------------
__global__ void bn_relu_nchw_k(const float* __restrict__ src, float* __restrict__ dst,
                               const float* __restrict__ stats,
                               int C, int dstCtot, int dstC0, int HW, int Bn) {
  long i = (long)blockIdx.x * 256 + threadIdx.x;
  const long tot = (long)Bn * C * HW;
  if (i >= tot) return;
  const int hw = (int)(i % HW);
  const long t = i / HW;
  const int c  = (int)(t % C);
  const int b  = (int)(t / C);
  float v = (src[i] - stats[2 * c]) * stats[2 * c + 1];
  v = v > 0.f ? v : 0.f;
  dst[((long)b * dstCtot + dstC0 + c) * HW + hw] = v;
}

// ---------------- dst_slice = x_slice * sigmoid(g) --------------------------
__global__ void gate_sigmoid_mul_k(const float* __restrict__ xsrc, int xCtot, int xc0,
                                   const float* __restrict__ g,
                                   float* __restrict__ dst, int dstCtot, int dstC0,
                                   int C, int HW, int Bn) {
  long i = (long)blockIdx.x * 256 + threadIdx.x;
  const long tot = (long)Bn * C * HW;
  if (i >= tot) return;
  const int hw = (int)(i % HW);
  const long t = i / HW;
  const int c  = (int)(t % C);
  const int b  = (int)(t / C);
  const float xv = xsrc[((long)b * xCtot + xc0 + c) * HW + hw];
  dst[((long)b * dstCtot + dstC0 + c) * HW + hw] = xv * sigmoidf_(g[i]);
}

// ---------------- 3x3 max pool, stride 1, pad 1 (from channel slice) --------
__global__ void maxpool3_k(const float* __restrict__ src, int sCtot, int sc0,
                           float* __restrict__ dst, int C, int Hh, int Ww, int Bn) {
  long i = (long)blockIdx.x * 256 + threadIdx.x;
  const long tot = (long)Bn * C * Hh * Ww;
  if (i >= tot) return;
  const int w = (int)(i % Ww);
  long t = i / Ww;
  const int h = (int)(t % Hh); t /= Hh;
  const int c = (int)(t % C);
  const int b = (int)(t / C);
  const float* s = src + ((long)b * sCtot + sc0 + c) * Hh * Ww;
  float m = -3.4e38f;
  for (int dh = -1; dh <= 1; ++dh) {
    const int hh = h + dh; if (hh < 0 || hh >= Hh) continue;
    for (int dw = -1; dw <= 1; ++dw) {
      const int ww2 = w + dw; if (ww2 < 0 || ww2 >= Ww) continue;
      const float v = s[hh * Ww + ww2];
      m = v > m ? v : m;
    }
  }
  dst[i] = m;
}

// ---------------- BlurPool: reflect pad (1,2), 4x4 binomial, stride 3 -------
__global__ void blurpool_k(const float* __restrict__ src, float* __restrict__ dst,
                           int C, int Hin, int Hout, int Bn) {
  long i = (long)blockIdx.x * 256 + threadIdx.x;
  const long tot = (long)Bn * C * Hout * Hout;
  if (i >= tot) return;
  const int ow = (int)(i % Hout);
  long t = i / Hout;
  const int oh = (int)(t % Hout); t /= Hout;
  const int c  = (int)(t % C);
  const int b  = (int)(t / C);
  const float f[4] = {1.f, 3.f, 3.f, 1.f};
  const float* s = src + ((long)b * C + c) * Hin * Hin;
  float acc = 0.f;
  for (int ki = 0; ki < 4; ++ki) {
    int qh = oh * 3 + ki - 1;                 // reflect (no edge repeat)
    if (qh < 0) qh = -qh;
    if (qh >= Hin) qh = 2 * (Hin - 1) - qh;
    for (int kj = 0; kj < 4; ++kj) {
      int qw = ow * 3 + kj - 1;
      if (qw < 0) qw = -qw;
      if (qw >= Hin) qw = 2 * (Hin - 1) - qw;
      acc += f[ki] * f[kj] * s[qh * Hin + qw];
    }
  }
  dst[i] = acc * (1.f / 64.f);
}

// ---------------- generic depthwise conv (with dilation) --------------------
__global__ void dwconv_k(const float* __restrict__ src, float* __restrict__ dst,
                         const float* __restrict__ w, const float* __restrict__ bias,
                         int C, int Hh, int Ww, int KH, int KW,
                         int ph, int pw, int dil, int Bn) {
  long i = (long)blockIdx.x * 256 + threadIdx.x;
  const long tot = (long)Bn * C * Hh * Ww;
  if (i >= tot) return;
  const int x = (int)(i % Ww);
  long t = i / Ww;
  const int y = (int)(t % Hh); t /= Hh;
  const int c = (int)(t % C);
  const int b = (int)(t / C);
  const float* s  = src + ((long)b * C + c) * Hh * Ww;
  const float* wk = w + (long)c * KH * KW;
  float acc = bias ? bias[c] : 0.f;
  for (int kh = 0; kh < KH; ++kh) {
    const int ih = y + kh * dil - ph;
    if (ih < 0 || ih >= Hh) continue;
    for (int kw = 0; kw < KW; ++kw) {
      const int iw = x + kw * dil - pw;
      if (iw < 0 || iw >= Ww) continue;
      acc += wk[kh * KW + kw] * s[ih * Ww + iw];
    }
  }
  dst[i] = acc;
}

// ---------------- shear transforms (S=22) -----------------------------------
__global__ void h_tf_k(const float* __restrict__ src, float* __restrict__ dst,
                       int C, int S, int Bn) {           // dst (B,C,S,2S-1)
  const int W2 = 2 * S - 1;
  long i = (long)blockIdx.x * 256 + threadIdx.x;
  const long tot = (long)Bn * C * S * W2;
  if (i >= tot) return;
  const int cc = (int)(i % W2);
  long t = i / W2;
  const int r = (int)(t % S); t /= S;
  const int c = (int)(t % C);
  const int b = (int)(t / C);
  const int j  = r * W2 + cc;
  const int pr = j / (2 * S), pc = j % (2 * S);
  dst[i] = (pc < S) ? src[(((long)b * C + c) * S + pr) * S + pc] : 0.f;
}

__global__ void inv_h_tf_k(const float* __restrict__ src, float* __restrict__ dst,
                           int C, int S, int Bn) {       // src (B,C,S,2S-1) -> (B,C,S,S)
  const int W2 = 2 * S - 1;
  long i = (long)blockIdx.x * 256 + threadIdx.x;
  const long tot = (long)Bn * C * S * S;
  if (i >= tot) return;
  const int cc = (int)(i % S);
  long t = i / S;
  const int r = (int)(t % S); t /= S;
  const int c = (int)(t % C);
  const int b = (int)(t / C);
  const int j = r * 2 * S + cc;
  dst[i] = src[(((long)b * C + c) * S + j / W2) * W2 + (j % W2)];
}

__global__ void v_tf_k(const float* __restrict__ src, float* __restrict__ dst,
                       int C, int S, int Bn) {           // dst (B,C,2S-1,S)
  const int W2 = 2 * S - 1;
  long i = (long)blockIdx.x * 256 + threadIdx.x;
  const long tot = (long)Bn * C * W2 * S;
  if (i >= tot) return;
  const int q = (int)(i % S);
  long t = i / S;
  const int p = (int)(t % W2); t /= W2;
  const int c = (int)(t % C);
  const int b = (int)(t / C);
  const int j  = q * W2 + p;
  const int pr = j / (2 * S), pc = j % (2 * S);
  dst[i] = (pc < S) ? src[(((long)b * C + c) * S + pc) * S + pr] : 0.f;
}

__global__ void inv_v_tf_k(const float* __restrict__ src, float* __restrict__ dst,
                           int C, int S, int Bn) {       // src (B,C,2S-1,S) -> (B,C,S,S)
  const int W2 = 2 * S - 1;
  long i = (long)blockIdx.x * 256 + threadIdx.x;
  const long tot = (long)Bn * C * S * S;
  if (i >= tot) return;
  const int cc = (int)(i % S);
  long t = i / S;
  const int r = (int)(t % S); t /= S;
  const int c = (int)(t % C);
  const int b = (int)(t / C);
  const int j = cc * 2 * S + r;
  dst[i] = src[(((long)b * C + c) * W2 + (j % W2)) * S + (j / W2)];
}

__global__ void add4_k(const float* a, const float* b2, const float* c,
                       const float* d, float* o, long n) {
  long i = (long)blockIdx.x * 256 + threadIdx.x;
  if (i < n) o[i] = a[i] + b2[i] + c[i] + d[i];
}

// ---------------- MRA nearest-upsampled sigmoid gate ------------------------
__global__ void mra_gate_k(const float* __restrict__ xsrc, int xCtot, int xc0,
                           const float* __restrict__ s, const float* __restrict__ stats,
                           float* __restrict__ dst, int dstCtot, int dstC0,
                           int C, int Hh, int Ww, int GS, int Bn) {
  long i = (long)blockIdx.x * 256 + threadIdx.x;
  const long tot = (long)Bn * C * Hh * Ww;
  if (i >= tot) return;
  const int w = (int)(i % Ww);
  long t = i / Ww;
  const int h = (int)(t % Hh); t /= Hh;
  const int c = (int)(t % C);
  const int b = (int)(t / C);
  const int ri = (h * GS) / Hh, ci = (w * GS) / Ww;
  const float sv = s[(((long)b * C + c) * GS + ri) * GS + ci];
  const float g  = sigmoidf_((sv - stats[2 * c]) * stats[2 * c + 1]);
  const long hw  = (long)h * Ww + w;
  dst[((long)b * dstCtot + dstC0 + c) * Hh * Ww + hw] =
      xsrc[((long)b * xCtot + xc0 + c) * Hh * Ww + hw] * g;
}

// ---------------- 2x2 max pool with argmax slot -----------------------------
__global__ void maxpool2_k(const float* __restrict__ src, int sCtot, int sc0,
                           float* __restrict__ dst, int* __restrict__ idx,
                           int C, int Hout, int Wout, int Bn) {
  long i = (long)blockIdx.x * 256 + threadIdx.x;
  const long tot = (long)Bn * C * Hout * Wout;
  if (i >= tot) return;
  const int w = (int)(i % Wout);
  long t = i / Wout;
  const int h = (int)(t % Hout); t /= Hout;
  const int c = (int)(t % C);
  const int b = (int)(t / C);
  const int Hin = 2 * Hout, Win = 2 * Wout;
  const float* s = src + (((long)b * sCtot + sc0 + c) * Hin + 2 * h) * Win + 2 * w;
  const float v0 = s[0], v1 = s[1], v2 = s[Win], v3 = s[Win + 1];
  float best = v0; int bi = 0;
  if (v1 > best) { best = v1; bi = 1; }
  if (v2 > best) { best = v2; bi = 2; }
  if (v3 > best) { best = v3; bi = 3; }
  dst[i] = best;
  idx[i] = bi;
}

// ---------------- 2x2 max unpool (optional fused BN), writes all 4 slots ----
__global__ void bn_unpool_k(const float* __restrict__ src, const int* __restrict__ idx,
                            const float* __restrict__ stats,
                            float* __restrict__ dst, int dstCtot, int dstC0,
                            int C, int Hs, int Ws, int Bn) {
  long i = (long)blockIdx.x * 256 + threadIdx.x;
  const long tot = (long)Bn * C * Hs * Ws;
  if (i >= tot) return;
  const int w = (int)(i % Ws);
  long t = i / Ws;
  const int h = (int)(t % Hs); t /= Hs;
  const int c = (int)(t % C);
  const int b = (int)(t / C);
  float v = src[i];
  if (stats) v = (v - stats[2 * c]) * stats[2 * c + 1];
  const int slot = idx[i];
  const int Ho = 2 * Hs, Wo = 2 * Ws;
  float* d = dst + (((long)b * dstCtot + dstC0 + c) * Ho + 2 * h) * Wo + 2 * w;
  d[0]      = (slot == 0) ? v : 0.f;
  d[1]      = (slot == 1) ? v : 0.f;
  d[Wo]     = (slot == 2) ? v : 0.f;
  d[Wo + 1] = (slot == 3) ? v : 0.f;
}

// ---------------- channel mean/max over concat(a1s,a2s) ---------------------
__global__ void aggregate_k(const float* __restrict__ a1s, const float* __restrict__ a2s,
                            float* __restrict__ agg, int Cc, int HW, int Bn) {
  long i = (long)blockIdx.x * 256 + threadIdx.x;
  const long tot = (long)Bn * HW;
  if (i >= tot) return;
  const int hw = (int)(i % HW);
  const int b  = (int)(i / HW);
  float s = 0.f, m = -3.4e38f;
  for (int c = 0; c < Cc; ++c) {
    const float v = a1s[((long)b * Cc + c) * HW + hw];
    s += v; m = v > m ? v : m;
  }
  for (int c = 0; c < Cc; ++c) {
    const float v = a2s[((long)b * Cc + c) * HW + hw];
    s += v; m = v > m ? v : m;
  }
  agg[((long)b * 2 + 0) * HW + hw] = s / (float)(2 * Cc);
  agg[((long)b * 2 + 1) * HW + hw] = m;
}

// ---------------- 2->2 ch 7x7 conv pad 3 + bias + sigmoid -------------------
__global__ void sqconv_k(const float* __restrict__ agg, const float* __restrict__ w,
                         const float* __restrict__ bias, float* __restrict__ sig,
                         int Hh, int Ww, int Bn) {
  long i = (long)blockIdx.x * 256 + threadIdx.x;
  const long tot = (long)Bn * 2 * Hh * Ww;
  if (i >= tot) return;
  const int x = (int)(i % Ww);
  long t = i / Ww;
  const int y  = (int)(t % Hh); t /= Hh;
  const int oc = (int)(t % 2);
  const int b  = (int)(t / 2);
  float acc = bias[oc];
  for (int ic = 0; ic < 2; ++ic) {
    const float* s  = agg + ((long)b * 2 + ic) * Hh * Ww;
    const float* wk = w + ((long)oc * 2 + ic) * 49;
    for (int kh = 0; kh < 7; ++kh) {
      const int ih = y + kh - 3; if (ih < 0 || ih >= Hh) continue;
      for (int kw = 0; kw < 7; ++kw) {
        const int iw = x + kw - 3; if (iw < 0 || iw >= Ww) continue;
        acc += wk[kh * 7 + kw] * s[ih * Ww + iw];
      }
    }
  }
  sig[i] = sigmoidf_(acc);
}

// ---------------- attn mix: a1s*sig0 + a2s*sig1 ------------------------------
__global__ void attmix_k(const float* __restrict__ a1s, const float* __restrict__ a2s,
                         const float* __restrict__ sig, float* __restrict__ out,
                         int Cc, int HW, int Bn) {
  long i = (long)blockIdx.x * 256 + threadIdx.x;
  const long tot = (long)Bn * Cc * HW;
  if (i >= tot) return;
  const int hw = (int)(i % HW);
  const int b  = (int)(i / ((long)Cc * HW));
  out[i] = a1s[i] * sig[((long)b * 2 + 0) * HW + hw] +
           a2s[i] * sig[((long)b * 2 + 1) * HW + hw];
}

__global__ void mul_k(const float* a, const float* b2, float* o, long n) {
  long i = (long)blockIdx.x * 256 + threadIdx.x;
  if (i < n) o[i] = a[i] * b2[i];
}

// ===========================================================================

static inline unsigned nb(long tot) { return (unsigned)((tot + 255) / 256); }

extern "C" void kernel_launch(void* const* d_in, const int* in_sizes, int n_in,
                              void* d_out, int out_size, void* d_ws, size_t ws_size,
                              hipStream_t stream) {
  (void)in_sizes; (void)n_in; (void)out_size; (void)ws_size;
  constexpr int Bn = 32, DIM = 256, Hh = 64, Ww = 64, HW = 64 * 64;
  constexpr int GS = 22;

  const float* x        = (const float*)d_in[0];
  const float* pa_w1    = (const float*)d_in[1];
  const float* pa_w2    = (const float*)d_in[2];
  const float* la_w     = (const float*)d_in[3];
  const float* mra_h1   = (const float*)d_in[4];
  const float* mra_v1   = (const float*)d_in[5];
  const float* mra_h2   = (const float*)d_in[6];
  const float* mra_v2   = (const float*)d_in[7];
  const float* gp1_w    = (const float*)d_in[8];
  const float* gp1_b    = (const float*)d_in[9];
  const float* gc0_w    = (const float*)d_in[10];
  const float* gc0_b    = (const float*)d_in[11];
  const float* gsp_w    = (const float*)d_in[12];
  const float* gsp_b    = (const float*)d_in[13];
  const float* gc1_w    = (const float*)d_in[14];
  const float* gc1_b    = (const float*)d_in[15];
  const float* gc2_w    = (const float*)d_in[16];
  const float* gc2_b    = (const float*)d_in[17];
  const float* gc_w     = (const float*)d_in[18];
  const float* gc_b     = (const float*)d_in[19];
  const float* gsq_w    = (const float*)d_in[20];
  const float* gsq_b    = (const float*)d_in[21];
  const float* gp2_w    = (const float*)d_in[22];
  const float* gp2_b    = (const float*)d_in[23];
  const float* mlp_w1   = (const float*)d_in[24];
  const float* mlp_w2   = (const float*)d_in[25];
  float* out = (float*)d_out;

  char* P = (char*)d_ws;
  auto alloc = [&](size_t bytes) -> void* {
    void* p = (void*)P;
    P += (bytes + 255) & ~(size_t)255;
    return p;
  };

  // ---------- persistent: zero page, bf16 weights (Mpad rows), stats, x_att
  __bf16* zpage = (__bf16*)alloc(4096);
  __bf16* w_pa1 = (__bf16*)alloc((size_t)256 * 64 * 2);
  __bf16* w_pa2 = (__bf16*)alloc((size_t)64 * 256 * 2);
  __bf16* w_la  = (__bf16*)alloc((size_t)9 * 64 * 64 * 2);
  __bf16* w_gp1 = (__bf16*)alloc((size_t)64 * 64 * 2);
  __bf16* w_gc1 = (__bf16*)alloc((size_t)64 * 64 * 2);   // M 32 -> pad 64
  __bf16* w_gc2 = (__bf16*)alloc((size_t)64 * 64 * 2);   // M 32 -> pad 64
  __bf16* w_gc  = (__bf16*)alloc((size_t)64 * 32 * 2);
  __bf16* w_gp2 = (__bf16*)alloc((size_t)64 * 64 * 2);
  __bf16* w_m1  = (__bf16*)alloc((size_t)512 * 256 * 2);
  __bf16* w_m2  = (__bf16*)alloc((size_t)256 * 512 * 2);
  float* st_pa  = (float*)alloc(256 * 2 * 4);
  float* st_la  = (float*)alloc(64 * 2 * 4);
  float* st_mra = (float*)alloc(64 * 2 * 4);
  float* st_ga  = (float*)alloc(64 * 2 * 4);
  float* st_mlp = (float*)alloc(512 * 2 * 4);
  float* x_att  = (float*)alloc((size_t)Bn * DIM * HW * 4);   // 134 MB

  zero_k<<<nb(2048), 256, 0, stream>>>(zpage, 2048);
  pack_w<<<nb(256 * 64), 256, 0, stream>>>(pa_w1, w_pa1, 256, 256, 64, 1, 1);
  pack_w<<<nb(64 * 256), 256, 0, stream>>>(pa_w2, w_pa2, 64, 64, 256, 1, 1);
  pack_w<<<nb(9 * 64 * 64), 256, 0, stream>>>(la_w, w_la, 64, 64, 64, 3, 3);
  pack_w<<<nb(64 * 64), 256, 0, stream>>>(gp1_w, w_gp1, 64, 64, 64, 1, 1);
  pack_w<<<nb(64 * 64), 256, 0, stream>>>(gc1_w, w_gc1, 32, 64, 64, 1, 1);
  pack_w<<<nb(64 * 64), 256, 0, stream>>>(gc2_w, w_gc2, 32, 64, 64, 1, 1);
  pack_w<<<nb(64 * 32), 256, 0, stream>>>(gc_w, w_gc, 64, 64, 32, 1, 1);
  pack_w<<<nb(64 * 64), 256, 0, stream>>>(gp2_w, w_gp2, 64, 64, 64, 1, 1);
  pack_w<<<nb(512 * 256), 256, 0, stream>>>(mlp_w1, w_m1, 512, 512, 256, 1, 1);
  pack_w<<<nb(256 * 512), 256, 0, stream>>>(mlp_w2, w_m2, 256, 256, 512, 1, 1);

  char* S0 = P;   // scratch arena start (reset per phase)

  // =================== Branch 1: PA (pixel attention) ===================
  {
    const long tot64  = (long)Bn * HW * 64;
    const long tot256 = (long)Bn * HW * 256;
    __bf16* xb1 = (__bf16*)alloc((size_t)tot64 * 2);
    to_nhwc_bf16<<<dim3(HW / 32, 2, Bn), 256, 0, stream>>>(x, xb1, nullptr, 64, 256, 0, HW);
    float* t1 = (float*)alloc((size_t)tot256 * 4);
    wmma_conv_gemm<<<dim3(1024, 4), 256, 0, stream>>>(w_pa1, xb1, zpage, t1, nullptr, nullptr,
                                                      256, 256, 64, Hh, Ww, 1, 1, 0, 0, 0);
    bn_stats_k<<<256, 256, 0, stream>>>(t1, st_pa, 256, HW, Bn);
    __bf16* tb1 = (__bf16*)alloc((size_t)tot256 * 2);
    to_nhwc_bf16<<<dim3(HW / 32, 8, Bn), 256, 0, stream>>>(t1, tb1, st_pa, 256, 256, 0, HW);
    float* t2 = (float*)alloc((size_t)tot64 * 4);
    wmma_conv_gemm<<<dim3(1024, 1), 256, 0, stream>>>(w_pa2, tb1, zpage, t2, nullptr, nullptr,
                                                      64, 64, 256, Hh, Ww, 1, 1, 0, 0, 0);
    gate_sigmoid_mul_k<<<nb(tot64), 256, 0, stream>>>(x, 256, 0, t2, x_att, 256, 0,
                                                      64, HW, Bn);
  }

  // =================== Branch 2: LA (3x3 conv as 9 GEMM taps) ===========
  {
    P = S0;
    const long tot64 = (long)Bn * HW * 64;
    __bf16* xb2 = (__bf16*)alloc((size_t)tot64 * 2);
    to_nhwc_bf16<<<dim3(HW / 32, 2, Bn), 256, 0, stream>>>(x, xb2, nullptr, 64, 256, 64, HW);
    float* t3 = (float*)alloc((size_t)tot64 * 4);
    wmma_conv_gemm<<<dim3(1024, 1), 256, 0, stream>>>(w_la, xb2, zpage, t3, nullptr, nullptr,
                                                      64, 64, 64, Hh, Ww, 3, 3, 1, 1, 0);
    bn_stats_k<<<64, 256, 0, stream>>>(t3, st_la, 64, HW, Bn);
    bn_relu_nchw_k<<<nb(tot64), 256, 0, stream>>>(t3, x_att, st_la, 64, 256, 64, HW, Bn);
  }

  // =================== Branch 3: MRA ====================================
  {
    P = S0;
    const long totF  = (long)Bn * 64 * HW;
    const long tot22 = (long)Bn * 64 * GS * GS;
    const long totHT = (long)Bn * 64 * GS * (2 * GS - 1);
    float* m1 = (float*)alloc((size_t)totF * 4);
    maxpool3_k<<<nb(totF), 256, 0, stream>>>(x, 256, 128, m1, 64, Hh, Ww, Bn);
    float* m2 = (float*)alloc((size_t)tot22 * 4);
    blurpool_k<<<nb(tot22), 256, 0, stream>>>(m1, m2, 64, 64, GS, Bn);
    float* xh1 = (float*)alloc((size_t)tot22 * 4);
    dwconv_k<<<nb(tot22), 256, 0, stream>>>(m2, xh1, mra_h1, nullptr, 64, GS, GS,
                                            11, 3, 5, 1, 1, Bn);
    float* xw1 = (float*)alloc((size_t)tot22 * 4);
    dwconv_k<<<nb(tot22), 256, 0, stream>>>(m2, xw1, mra_v1, nullptr, 64, GS, GS,
                                            3, 11, 1, 5, 1, Bn);
    float* ht = (float*)alloc((size_t)totHT * 4);
    h_tf_k<<<nb(totHT), 256, 0, stream>>>(m2, ht, 64, GS, Bn);
    float* ht2 = (float*)alloc((size_t)totHT * 4);
    dwconv_k<<<nb(totHT), 256, 0, stream>>>(ht, ht2, mra_h2, nullptr, 64, GS, 2 * GS - 1,
                                            11, 3, 5, 1, 1, Bn);
    float* xh2 = (float*)alloc((size_t)tot22 * 4);
    inv_h_tf_k<<<nb(tot22), 256, 0, stream>>>(ht2, xh2, 64, GS, Bn);
    float* vt = (float*)alloc((size_t)totHT * 4);
    v_tf_k<<<nb(totHT), 256, 0, stream>>>(m2, vt, 64, GS, Bn);
    float* vt2 = (float*)alloc((size_t)totHT * 4);
    dwconv_k<<<nb(totHT), 256, 0, stream>>>(vt, vt2, mra_v2, nullptr, 64, 2 * GS - 1, GS,
                                            3, 11, 1, 5, 1, Bn);
    float* xw2 = (float*)alloc((size_t)tot22 * 4);
    inv_v_tf_k<<<nb(tot22), 256, 0, stream>>>(vt2, xw2, 64, GS, Bn);
    float* ssum = (float*)alloc((size_t)tot22 * 4);
    add4_k<<<nb(tot22), 256, 0, stream>>>(xh1, xw1, xh2, xw2, ssum, tot22);
    bn_stats_k<<<64, 256, 0, stream>>>(ssum, st_mra, 64, GS * GS, Bn);
    mra_gate_k<<<nb(totF), 256, 0, stream>>>(x, 256, 128, ssum, st_mra, x_att, 256, 128,
                                             64, Hh, Ww, GS, Bn);
  }

  // =================== Branch 4: D_GA (GA12) ============================
  {
    P = S0;
    const long tot32s = (long)Bn * 64 * 32 * 32;
    const long tot16s = (long)Bn * 64 * 16 * 16;
    const long tot16h = (long)Bn * 32 * 16 * 16;
    float* x4p  = (float*)alloc((size_t)tot32s * 4);
    int*   idx1 = (int*)alloc((size_t)tot32s * 4);
    maxpool2_k<<<nb(tot32s), 256, 0, stream>>>(x, 256, 192, x4p, idx1, 64, 32, 32, Bn);
    float* xp   = (float*)alloc((size_t)tot16s * 4);
    int*   idx2 = (int*)alloc((size_t)tot16s * 4);
    maxpool2_k<<<nb(tot16s), 256, 0, stream>>>(x4p, 64, 0, xp, idx2, 64, 16, 16, Bn);
    __bf16* xpb = (__bf16*)alloc((size_t)tot16s * 2);
    to_nhwc_bf16<<<dim3(8, 2, Bn), 256, 0, stream>>>(xp, xpb, nullptr, 64, 64, 0, 256);
    float* h0 = (float*)alloc((size_t)tot16s * 4);
    wmma_conv_gemm<<<dim3(64, 1), 256, 0, stream>>>(w_gp1, xpb, zpage, h0, gp1_b, nullptr,
                                                    64, 64, 64, 16, 16, 1, 1, 0, 0, 1);
    float* a1 = (float*)alloc((size_t)tot16s * 4);
    dwconv_k<<<nb(tot16s), 256, 0, stream>>>(h0, a1, gc0_w, gc0_b, 64, 16, 16,
                                             5, 5, 2, 2, 1, Bn);
    float* a2 = (float*)alloc((size_t)tot16s * 4);
    dwconv_k<<<nb(tot16s), 256, 0, stream>>>(a1, a2, gsp_w, gsp_b, 64, 16, 16,
                                             7, 7, 9, 9, 3, Bn);
    __bf16* a1b = (__bf16*)alloc((size_t)tot16s * 2);
    to_nhwc_bf16<<<dim3(8, 2, Bn), 256, 0, stream>>>(a1, a1b, nullptr, 64, 64, 0, 256);
    __bf16* a2b = (__bf16*)alloc((size_t)tot16s * 2);
    to_nhwc_bf16<<<dim3(8, 2, Bn), 256, 0, stream>>>(a2, a2b, nullptr, 64, 64, 0, 256);
    float* a1s = (float*)alloc((size_t)tot16h * 4);
    wmma_conv_gemm<<<dim3(64, 1), 256, 0, stream>>>(w_gc1, a1b, zpage, a1s, gc1_b, nullptr,
                                                    32, 64, 64, 16, 16, 1, 1, 0, 0, 0);
    float* a2s = (float*)alloc((size_t)tot16h * 4);
    wmma_conv_gemm<<<dim3(64, 1), 256, 0, stream>>>(w_gc2, a2b, zpage, a2s, gc2_b, nullptr,
                                                    32, 64, 64, 16, 16, 1, 1, 0, 0, 0);
    float* agg = (float*)alloc((size_t)Bn * 2 * 256 * 4);
    aggregate_k<<<nb((long)Bn * 256), 256, 0, stream>>>(a1s, a2s, agg, 32, 256, Bn);
    float* sig = (float*)alloc((size_t)Bn * 2 * 256 * 4);
    sqconv_k<<<nb((long)Bn * 2 * 256), 256, 0, stream>>>(agg, gsq_w, gsq_b, sig, 16, 16, Bn);
    float* ain = (float*)alloc((size_t)tot16h * 4);
    attmix_k<<<nb(tot16h), 256, 0, stream>>>(a1s, a2s, sig, ain, 32, 256, Bn);
    __bf16* ainb = (__bf16*)alloc((size_t)tot16h * 2);
    to_nhwc_bf16<<<dim3(8, 1, Bn), 256, 0, stream>>>(ain, ainb, nullptr, 32, 32, 0, 256);
    float* attn = (float*)alloc((size_t)tot16s * 4);
    wmma_conv_gemm<<<dim3(64, 1), 256, 0, stream>>>(w_gc, ainb, zpage, attn, gc_b, nullptr,
                                                    64, 64, 32, 16, 16, 1, 1, 0, 0, 0);
    float* hm = (float*)alloc((size_t)tot16s * 4);
    mul_k<<<nb(tot16s), 256, 0, stream>>>(h0, attn, hm, tot16s);
    __bf16* hmb = (__bf16*)alloc((size_t)tot16s * 2);
    to_nhwc_bf16<<<dim3(8, 2, Bn), 256, 0, stream>>>(hm, hmb, nullptr, 64, 64, 0, 256);
    float* g2 = (float*)alloc((size_t)tot16s * 4);
    wmma_conv_gemm<<<dim3(64, 1), 256, 0, stream>>>(w_gp2, hmb, zpage, g2, gp2_b, nullptr,
                                                    64, 64, 64, 16, 16, 1, 1, 0, 0, 0);
    float* xg = (float*)alloc((size_t)tot32s * 4);
    bn_unpool_k<<<nb(tot16s), 256, 0, stream>>>(g2, idx2, nullptr, xg, 64, 0,
                                                64, 16, 16, Bn);            // ga12 unpool
    bn_stats_k<<<64, 256, 0, stream>>>(xg, st_ga, 64, 1024, Bn);
    bn_unpool_k<<<nb(tot32s), 256, 0, stream>>>(xg, idx1, st_ga, x_att, 256, 192,
                                                64, 32, 32, Bn);            // bn + outer unpool
  }

  // =================== MLP + residual ===================================
  {
    P = S0;
    const long totA = (long)Bn * HW * 256;
    const long totH = (long)Bn * HW * 512;
    __bf16* xab = (__bf16*)alloc((size_t)totA * 2);
    to_nhwc_bf16<<<dim3(HW / 32, 8, Bn), 256, 0, stream>>>(x_att, xab, nullptr, 256, 256, 0, HW);
    float* hf = (float*)alloc((size_t)totH * 4);
    wmma_conv_gemm<<<dim3(1024, 8), 256, 0, stream>>>(w_m1, xab, zpage, hf, nullptr, nullptr,
                                                      512, 512, 256, Hh, Ww, 1, 1, 0, 0, 0);
    bn_stats_k<<<512, 256, 0, stream>>>(hf, st_mlp, 512, HW, Bn);
    __bf16* hb2 = (__bf16*)alloc((size_t)totH * 2);
    to_nhwc_bf16<<<dim3(HW / 32, 16, Bn), 256, 0, stream>>>(hf, hb2, st_mlp, 512, 512, 0, HW);
    // out = x + conv(relu(bn(hf)), mlp_w2)  -- residual fused into GEMM store
    wmma_conv_gemm<<<dim3(1024, 4), 256, 0, stream>>>(w_m2, hb2, zpage, out, nullptr, x,
                                                      256, 256, 512, Hh, Ww, 1, 1, 0, 0, 0);
  }
}
